// TensorFlowModel_23441931502259
// MI455X (gfx1250) — matrix operand order repaired and checked by reading, not compiled, not run on hardware
//
#include <hip/hip_runtime.h>
#include <hip/hip_bf16.h>
#include <math.h>
#include <stdint.h>

typedef __attribute__((ext_vector_type(16))) __bf16 bf16x16;
typedef __attribute__((ext_vector_type(8)))  __bf16 bf16x8;
typedef __attribute__((ext_vector_type(8)))  float  f32x8;

// ---------------- problem geometry ----------------
#define DK        4096                    // feature dim 64*64
#define NR        2176                    // padded rows of [buf(1000); x(1024); zero-pad]
#define NPAD      2176                    // padded distance-matrix width
#define BUFN      1000
#define KNN       50
#define KSPLIT    4                       // split-K factor (K quarter = 1024)
#define PLSZ      (1024u * NPAD)          // one partial-dot plane (f32 elements)

// ---------------- workspace layout (bytes) ----------------
#define BHI_OFF   0u                      // NR*DK bf16 hi parts
#define BLO_OFF   17825792u               // NR*DK bf16 lo parts
#define NB_OFF    35651584u               // NR f32 exact row norms
#define PLN_OFF   35660288u               // KSPLIT * 1024*NPAD f32 partial dots
#define P1_OFF    71311872u               // 1024*30*30*4 f32 pooled conv1
#define P2_OFF    86057472u               // 1024*1690 f32 pooled conv2 (flatten order)
#define H1_OFF    92979712u               // 1024*100 f32 dense1 output

// =====================================================================
// CDNA5 async-copy helpers (ASYNCcnt path)
// =====================================================================
__device__ __forceinline__ void async_b128(unsigned ldsaddr, const __bf16* g) {
    asm volatile("global_load_async_to_lds_b128 %0, %1, off"
                 :: "v"(ldsaddr), "v"(g) : "memory");
}
__device__ __forceinline__ void wait_async0() {
    asm volatile("s_wait_asynccnt 0x0" ::: "memory");
}

__device__ __forceinline__ f32x8 wmma_bf16(bf16x16 a, bf16x16 b, f32x8 c) {
    return __builtin_amdgcn_wmma_f32_16x16x32_bf16(
        false, a, false, b, (short)0, c, false, false);
}

// =====================================================================
// 1) Convert rows of [buf ; x ; 0] to split-bf16 (hi+lo) + exact fp32 norms
// =====================================================================
__global__ void bf_convert(const float* __restrict__ x,
                           const float* __restrict__ buf,
                           __bf16* __restrict__ bhi,
                           __bf16* __restrict__ blo,
                           float* __restrict__ nb) {
    const int r = blockIdx.x;             // 0..NR-1
    const int tid = threadIdx.x;          // 256 threads
    const float* src = nullptr;
    if (r < BUFN)             src = buf + (size_t)r * DK;
    else if (r < BUFN + 1024) src = x + (size_t)(r - BUFN) * DK;

    float sumsq = 0.f;
    for (int e = tid; e < DK; e += 256) {
        float v = src ? src[e] : 0.f;
        __bf16 hi = (__bf16)v;
        __bf16 lo = (__bf16)(v - (float)hi);
        bhi[(size_t)r * DK + e] = hi;
        blo[(size_t)r * DK + e] = lo;
        sumsq += v * v;
    }
    __shared__ float red[256];
    red[tid] = sumsq;
    __syncthreads();
    for (int s = 128; s > 0; s >>= 1) {
        if (tid < s) red[tid] += red[tid + s];
        __syncthreads();
    }
    if (tid == 0) nb[r] = red[0];
}

// =====================================================================
// 2) Banded dot-product GEMM, split-bf16 WMMA, split-K=4.
//    Workgroup = 64M x 128N macro tile, 8 waves (4 row-tiles x 2 col-halves),
//    each wave 16M x 64N (4 tiles), K-quarter = 1024 in 32-element chunks.
//    B (hi+lo) double-buffered in LDS via global_load_async_to_lds_b128;
//    A streamed from global per wave. Stores raw dot partials.
// =====================================================================
__global__ void __launch_bounds__(256) knn_gemm(const __bf16* __restrict__ bhi,
                                                const __bf16* __restrict__ blo,
                                                float* __restrict__ planes) {
    // LDS: 2 buffers x (hi plane 10240B + lo plane 10240B); row stride 80B
    __shared__ __align__(16) char ldsB[40960];

    const int tid  = threadIdx.x;
    const int lane = tid & 31;
    const int wv   = tid >> 5;            // wave 0..7
    const int bx   = blockIdx.x;          // 16 row-blocks * 9 col-blocks * 4 ksplit
    const int ks    = bx & 3;
    const int t2    = bx >> 2;
    const int blk_c = t2 % 9;
    const int blk_r = t2 / 9;
    const int col0  = blk_r * 64 + blk_c * 128;   // <= 2111 < NPAD
    const int kbase = ks * 1024;

    const int wr   = wv & 3;              // row-tile within block
    const int wc   = wv >> 2;             // col half (0/1)
    const int i0   = blk_r * 64 + wr * 16;
    const int mrow = lane & 15;
    const int half = lane >> 4;

    const unsigned ldsBase = (unsigned)(uintptr_t)(void*)&ldsB[0];

    // --- async staging plan: 4 x 16B granules per thread per chunk ---
    // chunk = 128 cols x 32 K x {hi,lo} = 1024 granules; granule g:
    //   plane = g>>9, row = (g&511)>>2, k-subchunk c8 = g&3
    const __bf16* gB[4];
    unsigned ldsOff[4];
#pragma unroll
    for (int q = 0; q < 4; ++q) {
        int g     = wv * 128 + q * 32 + lane;
        int plane = g >> 9;
        int g2    = g & 511;
        int row   = g2 >> 2;
        int c8    = g2 & 3;
        const __bf16* src = plane ? blo : bhi;
        gB[q]     = src + (size_t)(col0 + row) * DK + kbase + c8 * 8;
        ldsOff[q] = ldsBase + (unsigned)(plane * 10240 + row * 80 + c8 * 16);
    }

    // --- A operand (x rows live at +BUFN in the B arrays) ---
    const size_t aBase = (size_t)(BUFN + i0 + mrow) * DK + kbase + half * 8;
    const __bf16* aHiP = bhi + aBase;
    const __bf16* aLoP = blo + aBase;

    f32x8 acc[4] = {};

    // stage chunk 0 into buffer 0
#pragma unroll
    for (int q = 0; q < 4; ++q) async_b128(ldsOff[q], gB[q]);

    for (int c = 0; c < 32; ++c) {
        const int k0 = c * 32;
        const unsigned bufo = (c & 1) ? 20480u : 0u;

        wait_async0();                    // my granules of chunk c landed
        __syncthreads();                  // everyone's landed; prev buffer free

        if (c + 1 < 32) {                 // stage chunk c+1 into other buffer
            const unsigned nbufo = ((c + 1) & 1) ? 20480u : 0u;
#pragma unroll
            for (int q = 0; q < 4; ++q)
                async_b128(ldsOff[q] + nbufo, gB[q] + (k0 + 32));
        }

        // A from global (two 8-elem K chunks per lane, hi & lo planes)
        bf16x8 a0 = *(const bf16x8*)(aHiP + k0);
        bf16x8 a1 = *(const bf16x8*)(aHiP + k0 + 16);
        bf16x8 l0 = *(const bf16x8*)(aLoP + k0);
        bf16x8 l1 = *(const bf16x8*)(aLoP + k0 + 16);
        bf16x16 ahi = __builtin_shufflevector(a0, a1, 0,1,2,3,4,5,6,7,8,9,10,11,12,13,14,15);
        bf16x16 alo = __builtin_shufflevector(l0, l1, 0,1,2,3,4,5,6,7,8,9,10,11,12,13,14,15);
        __builtin_prefetch(aHiP + k0 + 64, 0, 3);   // global_prefetch_b8

        const char* bbase = (const char*)ldsB + bufo;
#pragma unroll
        for (int u = 0; u < 4; ++u) {
            int nloc = (wc * 4 + u) * 16 + mrow;
            const char* ph = bbase + nloc * 80 + half * 16;   // hi plane
            const char* pl = ph + 10240;                      // lo plane
            bf16x8 h0 = *(const bf16x8*)ph;
            bf16x8 h1 = *(const bf16x8*)(ph + 32);
            bf16x8 b0 = *(const bf16x8*)pl;
            bf16x8 b1 = *(const bf16x8*)(pl + 32);
            bf16x16 bh = __builtin_shufflevector(h0, h1, 0,1,2,3,4,5,6,7,8,9,10,11,12,13,14,15);
            bf16x16 bl = __builtin_shufflevector(b0, b1, 0,1,2,3,4,5,6,7,8,9,10,11,12,13,14,15);
            acc[u] = wmma_bf16(ahi, bh, acc[u]);
            acc[u] = wmma_bf16(ahi, bl, acc[u]);
            acc[u] = wmma_bf16(alo, bh, acc[u]);
        }
    }

    // epilogue: store raw dot partials for this K-quarter
    float* pln = planes + (size_t)ks * PLSZ;
#pragma unroll
    for (int u = 0; u < 4; ++u) {
        int j = col0 + (wc * 4 + u) * 16 + mrow;
#pragma unroll
        for (int v = 0; v < 8; ++v) {
            int i = i0 + half * 8 + v;
            pln[(size_t)i * NPAD + j] = acc[u][v];
        }
    }
}

// =====================================================================
// 3) Per-row: sum K-partials, apply norms, bitonic-sort window,
//    mean of 50 smallest sqrt distances
// =====================================================================
__global__ void knn_select(const float* __restrict__ planes,
                           const float* __restrict__ nb,
                           float* __restrict__ out) {
    const int i = blockIdx.x;             // row 0..1023
    const int tid = threadIdx.x;          // 256 threads
    __shared__ float s[1024];

    const float nxi = nb[BUFN + i];
    for (int t = tid; t < 1024; t += 256) {
        if (t < BUFN) {
            int j = i + t;
            size_t o = (size_t)i * NPAD + j;
            float dot = planes[o] + planes[PLSZ + o]
                      + planes[2 * (size_t)PLSZ + o] + planes[3 * (size_t)PLSZ + o];
            s[t] = nxi + nb[j] - 2.0f * dot;
        } else {
            s[t] = 3.0e38f;
        }
    }
    __syncthreads();

    for (int k = 2; k <= 1024; k <<= 1) {
        for (int j = k >> 1; j > 0; j >>= 1) {
            for (int l = tid; l < 1024; l += 256) {
                int ixj = l ^ j;
                if (ixj > l) {
                    bool up = ((l & k) == 0);
                    float a = s[l], b = s[ixj];
                    if ((a > b) == up) { s[l] = b; s[ixj] = a; }
                }
            }
            __syncthreads();
        }
    }

    if (tid == 0) {
        float sum = 0.f;
        for (int t = 0; t < KNN; ++t)
            sum += sqrtf(fmaxf(s[t], 0.f));
        out[10240 + i] = sum / (float)KNN;
    }
}

// =====================================================================
// 4) CNN: fused conv1(5x5,1->4)+relu+maxpool2  ->  [1024,30,30,4]
// =====================================================================
__global__ void conv1pool(const float* __restrict__ xin,
                          const float* __restrict__ w1,
                          const float* __restrict__ b1,
                          float* __restrict__ pool1) {
    int idx = blockIdx.x * blockDim.x + threadIdx.x;
    if (idx >= 1024 * 30 * 30) return;
    int b  = idx / 900;
    int r  = idx % 900;
    int py = r / 30, px = r % 30;

    float m[4] = {-3.0e38f, -3.0e38f, -3.0e38f, -3.0e38f};
    for (int dy = 0; dy < 2; ++dy)
        for (int dx = 0; dx < 2; ++dx) {
            int y = py * 2 + dy, xx = px * 2 + dx;
            float s[4] = {b1[0], b1[1], b1[2], b1[3]};
#pragma unroll
            for (int ky = 0; ky < 5; ++ky)
#pragma unroll
                for (int kx = 0; kx < 5; ++kx) {
                    float v = xin[((size_t)b * 64 + y + ky) * 64 + (xx + kx)];
                    const float* wp = w1 + (ky * 5 + kx) * 4;
#pragma unroll
                    for (int c = 0; c < 4; ++c) s[c] += v * wp[c];
                }
#pragma unroll
            for (int c = 0; c < 4; ++c) m[c] = fmaxf(m[c], s[c]);
        }
#pragma unroll
    for (int c = 0; c < 4; ++c)
        pool1[(size_t)idx * 4 + c] = fmaxf(m[c], 0.f);
}

// =====================================================================
// 5) CNN: fused conv2(5x5,4->10)+relu+maxpool2 -> flatten [1024,1690]
// =====================================================================
__global__ void conv2pool(const float* __restrict__ pool1,
                          const float* __restrict__ w2,
                          const float* __restrict__ b2,
                          float* __restrict__ pool2) {
    int idx = blockIdx.x * blockDim.x + threadIdx.x;
    if (idx >= 1024 * 169) return;
    int b  = idx / 169;
    int r  = idx % 169;
    int py = r / 13, px = r % 13;

    float m[10];
#pragma unroll
    for (int c = 0; c < 10; ++c) m[c] = -3.0e38f;

    for (int dy = 0; dy < 2; ++dy)
        for (int dx = 0; dx < 2; ++dx) {
            int y = py * 2 + dy, xx = px * 2 + dx;
            float s[10];
#pragma unroll
            for (int c = 0; c < 10; ++c) s[c] = b2[c];
            for (int ky = 0; ky < 5; ++ky)
                for (int kx = 0; kx < 5; ++kx) {
                    const float* ip = pool1 + ((size_t)(b * 30 + y + ky) * 30 + (xx + kx)) * 4;
#pragma unroll
                    for (int ci = 0; ci < 4; ++ci) {
                        float v = ip[ci];
                        const float* wp = w2 + ((ky * 5 + kx) * 4 + ci) * 10;
#pragma unroll
                        for (int co = 0; co < 10; ++co) s[co] += v * wp[co];
                    }
                }
#pragma unroll
            for (int c = 0; c < 10; ++c) m[c] = fmaxf(m[c], s[c]);
        }
#pragma unroll
    for (int c = 0; c < 10; ++c)
        pool2[(size_t)b * 1690 + (py * 13 + px) * 10 + c] = fmaxf(m[c], 0.f);
}

// =====================================================================
// 6) dense1 (1690 -> 100) + relu, LDS-staged activation row
// =====================================================================
__global__ void dense1(const float* __restrict__ pool2,
                       const float* __restrict__ dw,
                       const float* __restrict__ db,
                       float* __restrict__ h1) {
    const int b = blockIdx.x;
    const int tid = threadIdx.x;          // 128 threads
    __shared__ float xrow[1690];
    for (int f = tid; f < 1690; f += 128)
        xrow[f] = pool2[(size_t)b * 1690 + f];
    __syncthreads();
    if (tid < 100) {
        float acc = db[tid];
        for (int f = 0; f < 1690; ++f)
            acc += xrow[f] * dw[(size_t)f * 100 + tid];
        h1[(size_t)b * 100 + tid] = fmaxf(acc, 0.f);
    }
}

// =====================================================================
// 7) logits (100 -> 10) + softmax
// =====================================================================
__global__ void logits_softmax(const float* __restrict__ h1,
                               const float* __restrict__ lw,
                               const float* __restrict__ lb,
                               float* __restrict__ out) {
    const int b = blockIdx.x;
    const int tid = threadIdx.x;          // 32 threads
    __shared__ float hrow[100];
    __shared__ float lg[10];
    for (int f = tid; f < 100; f += 32)
        hrow[f] = h1[(size_t)b * 100 + f];
    __syncthreads();
    if (tid < 10) {
        float acc = lb[tid];
        for (int f = 0; f < 100; ++f)
            acc += hrow[f] * lw[f * 10 + tid];
        lg[tid] = acc;
    }
    __syncthreads();
    if (tid == 0) {
        float mx = lg[0];
        for (int o = 1; o < 10; ++o) mx = fmaxf(mx, lg[o]);
        float sum = 0.f, e[10];
        for (int o = 0; o < 10; ++o) { e[o] = expf(lg[o] - mx); sum += e[o]; }
        float inv = 1.f / sum;
        for (int o = 0; o < 10; ++o) out[(size_t)b * 10 + o] = e[o] * inv;
    }
}

// =====================================================================
extern "C" void kernel_launch(void* const* d_in, const int* in_sizes, int n_in,
                              void* d_out, int out_size, void* d_ws, size_t ws_size,
                              hipStream_t stream) {
    const float* x   = (const float*)d_in[0];   // [1024,64,64,1]
    const float* buf = (const float*)d_in[1];   // [1000,64,64,1]
    const float* w1  = (const float*)d_in[2];   // [5,5,1,4]
    const float* b1  = (const float*)d_in[3];   // [4]
    const float* w2  = (const float*)d_in[4];   // [5,5,4,10]
    const float* b2  = (const float*)d_in[5];   // [10]
    const float* dw  = (const float*)d_in[6];   // [1690,100]
    const float* db  = (const float*)d_in[7];   // [100]
    const float* lw  = (const float*)d_in[8];   // [100,10]
    const float* lb  = (const float*)d_in[9];   // [10]
    float* out = (float*)d_out;                 // [1024*10 probs][1024 avg_dists]

    char* ws = (char*)d_ws;
    __bf16* bhi   = (__bf16*)(ws + BHI_OFF);
    __bf16* blo   = (__bf16*)(ws + BLO_OFF);
    float* nb     = (float*)(ws + NB_OFF);
    float* planes = (float*)(ws + PLN_OFF);
    float* pool1  = (float*)(ws + P1_OFF);
    float* pool2  = (float*)(ws + P2_OFF);
    float* h1     = (float*)(ws + H1_OFF);

    // kNN branch (WMMA + async-LDS path)
    bf_convert<<<NR, 256, 0, stream>>>(x, buf, bhi, blo, nb);
    knn_gemm<<<16 * 9 * KSPLIT, 256, 0, stream>>>(bhi, blo, planes);
    knn_select<<<1024, 256, 0, stream>>>(planes, nb, out);

    // CNN branch
    conv1pool<<<(1024 * 30 * 30 + 255) / 256, 256, 0, stream>>>(x, w1, b1, pool1);
    conv2pool<<<(1024 * 169 + 255) / 256, 256, 0, stream>>>(pool1, w2, b2, pool2);
    dense1<<<1024, 128, 0, stream>>>(pool2, dw, db, h1);
    logits_softmax<<<1024, 32, 0, stream>>>(h1, lw, lb, out);
}
